// VisionLinearAttention_60559038873832
// MI455X (gfx1250) — matrix operand order repaired and checked by reading, not compiled, not run on hardware
//
#include <hip/hip_runtime.h>

typedef __attribute__((ext_vector_type(16))) __bf16   v16bf;
typedef __attribute__((ext_vector_type(8)))  float    v8f;
typedef __attribute__((ext_vector_type(8)))  unsigned v8u;

// ---------------- helpers ----------------
__device__ __forceinline__ unsigned short bf16h(float f) {
    unsigned u = __float_as_uint(f);
    return (unsigned short)((u + 0x7FFFu + ((u >> 16) & 1u)) >> 16);
}
__device__ __forceinline__ unsigned bf16pack(float a, float b) {
    return (unsigned)bf16h(a) | ((unsigned)bf16h(b) << 16);
}
__device__ __forceinline__ void pack8(unsigned* dst, const float* src) {
    float4 f0 = *(const float4*)(src + 0);
    float4 f1 = *(const float4*)(src + 4);
    float4 f2 = *(const float4*)(src + 8);
    float4 f3 = *(const float4*)(src + 12);
    dst[0] = bf16pack(f0.x, f0.y); dst[1] = bf16pack(f0.z, f0.w);
    dst[2] = bf16pack(f1.x, f1.y); dst[3] = bf16pack(f1.z, f1.w);
    dst[4] = bf16pack(f2.x, f2.y); dst[5] = bf16pack(f2.z, f2.w);
    dst[6] = bf16pack(f3.x, f3.y); dst[7] = bf16pack(f3.z, f3.w);
}
__device__ __forceinline__ v8f wmma_bf16(v16bf a, v16bf b, v8f c) {
    return __builtin_amdgcn_wmma_f32_16x16x32_bf16(false, a, false, b,
                                                   (short)0, c, false, false);
}

// LDS-aperture flat address truncates to LDS byte offset (ISA 10.2)
__device__ __forceinline__ unsigned lds_off(const void* p) {
    return (unsigned)(unsigned long long)p;
}
// async DMA: 16 bytes per lane, global -> LDS, tracked by ASYNCcnt
__device__ __forceinline__ void async_ld128(const void* lds, const void* g) {
    asm volatile("global_load_async_to_lds_b128 %0, %1, off"
                 :: "v"(lds_off(lds)), "v"(g) : "memory");
}
__device__ __forceinline__ void wait_async0() {
    asm volatile("s_wait_asynccnt 0" ::: "memory");
}

// ---------------- fp32 -> packed bf16 conversion (8 elems / thread) ------------
__global__ __launch_bounds__(256) void to_bf16(const float* __restrict__ src,
                                               uint4* __restrict__ dst) {
    size_t g = (size_t)blockIdx.x * 256 + threadIdx.x;
    float4 f0 = ((const float4*)src)[2 * g];
    float4 f1 = ((const float4*)src)[2 * g + 1];
    uint4 o;
    o.x = bf16pack(f0.x, f0.y); o.y = bf16pack(f0.z, f0.w);
    o.z = bf16pack(f1.x, f1.y); o.w = bf16pack(f1.z, f1.w);
    dst[g] = o;
}

// ---------------- sinusoidal PE table [1024][1024] ----------------
__global__ __launch_bounds__(256) void pe_kernel(float* __restrict__ pe) {
    int idx = blockIdx.x * 256 + threadIdx.x;   // 1024*512 pairs
    int n  = idx >> 9;
    int i2 = idx & 511;
    float i   = (float)(2 * i2);
    float ang = (float)n * __expf(-(i * (1.0f / 1024.0f)) * 9.210340371976184f); // ln(1e4)
    pe[n * 1024 + 2 * i2]     = __sinf(ang);
    pe[n * 1024 + 2 * i2 + 1] = __cosf(ang);
}

// ---------------- patch-embed GEMM (bf16 inputs, async-LDS staged, dbl-buffered)
// T[m][d] = sum_k A[m][k] * W[d][k] + bias[d] + pe[tok][d]
// M = 16384 (b*1024 + token), K = 768 (c*256 + p*16 + q), Nd = 1024
__global__ __launch_bounds__(256) void embed_gemm(const unsigned short* __restrict__ xb,
                                                  const unsigned short* __restrict__ wb,
                                                  const float* __restrict__ bias,
                                                  const float* __restrict__ pe,
                                                  float* __restrict__ T) {
    __shared__ unsigned lA[2][128 * 16];   // per buf: 128 rows x 32 bf16
    __shared__ unsigned lB[2][128 * 16];   // per buf: 128 d-rows x 32 bf16

    const int tid  = threadIdx.x;
    const int wid  = tid >> 5;
    const int lane = tid & 31;
    const int mtile = blockIdx.x * 128;
    const int ntile = blockIdx.y * 128;
    const int wm = (wid >> 1) * 32;
    const int wn = (wid & 1) * 64;

    const int srow  = tid >> 1;
    const int shalf = tid & 1;
    const int m  = mtile + srow;
    const int b  = m >> 10;
    const int n  = m & 1023;
    const int hp = n >> 5, wp = n & 31;

    auto stage = [&](int k0, int buf) {
        int kk = k0 + shalf * 16;
        int c  = kk >> 8;
        int p  = (kk & 255) >> 4;
        const unsigned short* sa = xb + ((size_t)(b * 3 + c) * 512 + hp * 16 + p) * 512
                                      + wp * 16;
        unsigned* da = &lA[buf][srow * 16 + shalf * 8];
        async_ld128(da,     sa);
        async_ld128(da + 4, sa + 8);
        const unsigned short* sb = wb + (size_t)(ntile + srow) * 768 + k0 + shalf * 16;
        unsigned* db = &lB[buf][srow * 16 + shalf * 8];
        async_ld128(db,     sb);
        async_ld128(db + 4, sb + 8);
    };

    v8f acc[2][4] = {};
    stage(0, 0);

    for (int i = 0; i < 24; ++i) {           // K = 768 = 24 x 32
        const int buf = i & 1;
        wait_async0();                        // my buf-i DMAs landed
        __syncthreads();                      // everyone's landed
        if (i + 1 < 24) stage((i + 1) * 32, buf ^ 1);

        const int abase = (lane >> 4) << 2;   // 0 or 4
        v16bf af[2];
        for (int t = 0; t < 2; ++t) {
            const unsigned* pA = &lA[buf][(wm + t * 16 + (lane & 15)) * 16];
            v8u a;
            a[0] = pA[abase + 0]; a[1] = pA[abase + 1];
            a[2] = pA[abase + 2]; a[3] = pA[abase + 3];
            a[4] = pA[abase + 8]; a[5] = pA[abase + 9];
            a[6] = pA[abase + 10]; a[7] = pA[abase + 11];
            af[t] = __builtin_bit_cast(v16bf, a);
        }
        const int bbase = (lane >> 4) << 2;   // 0 or 4
        for (int u = 0; u < 4; ++u) {
            const unsigned* pB = &lB[buf][(wn + u * 16 + (lane & 15)) * 16];
            v8u bb;
            for (int j = 0; j < 8; ++j) bb[j] = pB[bbase + j + (j & 4)];
            v16bf bf = __builtin_bit_cast(v16bf, bb);
            acc[0][u] = wmma_bf16(af[0], bf, acc[0][u]);
            acc[1][u] = wmma_bf16(af[1], bf, acc[1][u]);
        }
        __syncthreads();                      // done reading buf before re-stage
    }

    for (int t = 0; t < 2; ++t) {
        int rbase = mtile + wm + t * 16 + ((lane >> 4) << 3);
        for (int u = 0; u < 4; ++u) {
            int col = ntile + wn + u * 16 + (lane & 15);
            float bv = bias[col];
            for (int r = 0; r < 8; ++r) {
                int row = rbase + r;
                int tok = row & 1023;
                T[(size_t)row * 1024 + col] = acc[t][u][r] + bv + pe[tok * 1024 + col];
            }
        }
    }
}

// ---------------- q softmax over head_dim (rows of 64, one wave per row) --------
__global__ __launch_bounds__(256) void qsoftmax(float* __restrict__ T) {
    int row  = blockIdx.x * 8 + (threadIdx.x >> 5);
    int lane = threadIdx.x & 31;
    float* p = T + (size_t)row * 64;
    float a = p[lane], b = p[lane + 32];
    float mx = fmaxf(a, b);
    for (int off = 16; off; off >>= 1) mx = fmaxf(mx, __shfl_xor(mx, off));
    float ea = __expf(a - mx), eb = __expf(b - mx);
    float s = ea + eb;
    for (int off = 16; off; off >>= 1) s += __shfl_xor(s, off);
    float inv = __frcp_rn(s);
    p[lane] = ea * inv; p[lane + 32] = eb * inv;
}

// ---------------- k softmax over tokens (columns, stride 1024) ------------------
__global__ __launch_bounds__(256) void ksoftmax(float* __restrict__ T) {
    int b = blockIdx.y;
    int d = blockIdx.x * 256 + threadIdx.x;
    float* p = T + (size_t)b * 1024 * 1024 + d;
    float mx = -3.4e38f;
    for (int n = 0; n < 1024; ++n) mx = fmaxf(mx, p[(size_t)n * 1024]);
    float s = 0.f;
    for (int n = 0; n < 1024; ++n) s += __expf(p[(size_t)n * 1024] - mx);
    float inv = __frcp_rn(s);
    for (int n = 0; n < 1024; ++n) {
        float v = p[(size_t)n * 1024];
        p[(size_t)n * 1024] = __expf(v - mx) * inv;
    }
}

// ---------------- scores[b][h] = k_sm^T (64x1024) @ v (1024x64) -> 64x64 --------
__global__ __launch_bounds__(128) void kv_scores(const float* __restrict__ K,
                                                 const float* __restrict__ V,
                                                 float* __restrict__ S) {
    __shared__ unsigned lK[64 * 16];   // transposed: [d][32 tokens] bf16
    __shared__ unsigned lV[64 * 16];   // transposed: [e][32 tokens] bf16
    int h = blockIdx.x, b = blockIdx.y;
    int tid = threadIdx.x, wid = tid >> 5, lane = tid & 31;
    const float* Kp = K + (size_t)b * 1024 * 1024 + h * 64;
    const float* Vp = V + (size_t)b * 1024 * 1024 + h * 64;

    int tok = tid >> 2;          // 0..31
    int dg  = (tid & 3) * 16;    // 0,16,32,48
    unsigned short* sK = (unsigned short*)lK;
    unsigned short* sV = (unsigned short*)lV;

    v8f acc[4] = {};
    for (int n0 = 0; n0 < 1024; n0 += 32) {
        const float* ks = Kp + (size_t)(n0 + tok) * 1024 + dg;
        const float* vs = Vp + (size_t)(n0 + tok) * 1024 + dg;
        for (int i = 0; i < 4; ++i) {
            float4 fk = *(const float4*)(ks + 4 * i);
            float4 fv = *(const float4*)(vs + 4 * i);
            sK[(dg + 4 * i + 0) * 32 + tok] = bf16h(fk.x);
            sK[(dg + 4 * i + 1) * 32 + tok] = bf16h(fk.y);
            sK[(dg + 4 * i + 2) * 32 + tok] = bf16h(fk.z);
            sK[(dg + 4 * i + 3) * 32 + tok] = bf16h(fk.w);
            sV[(dg + 4 * i + 0) * 32 + tok] = bf16h(fv.x);
            sV[(dg + 4 * i + 1) * 32 + tok] = bf16h(fv.y);
            sV[(dg + 4 * i + 2) * 32 + tok] = bf16h(fv.z);
            sV[(dg + 4 * i + 3) * 32 + tok] = bf16h(fv.w);
        }
        __syncthreads();
        int drow  = wid * 16 + (lane & 15);
        int abase = drow * 16 + ((lane >> 4) << 2);
        v8u a;
        a[0] = lK[abase + 0]; a[1] = lK[abase + 1];
        a[2] = lK[abase + 2]; a[3] = lK[abase + 3];
        a[4] = lK[abase + 8]; a[5] = lK[abase + 9];
        a[6] = lK[abase + 10]; a[7] = lK[abase + 11];
        v16bf af = __builtin_bit_cast(v16bf, a);
        for (int u = 0; u < 4; ++u) {
            int e = u * 16 + (lane & 15);
            int bbase = e * 16 + ((lane >> 4) << 2);
            v8u bb;
            for (int j = 0; j < 8; ++j) bb[j] = lV[bbase + j + (j & 4)];
            acc[u] = wmma_bf16(af, __builtin_bit_cast(v16bf, bb), acc[u]);
        }
        __syncthreads();
    }
    float* Sp = S + ((size_t)(b * 16 + h)) * 64 * 64;
    int rbase = wid * 16 + ((lane >> 4) << 3);
    for (int u = 0; u < 4; ++u) {
        int e = u * 16 + (lane & 15);
        for (int r = 0; r < 8; ++r)
            Sp[(size_t)(rbase + r) * 64 + e] = acc[u][r];
    }
}

// ---------------- ctx = q_sm (1024x64) @ scores (64x64) -> out[b][n][h*64+e] -----
__global__ __launch_bounds__(256) void ctx_gemm(const float* __restrict__ Q,
                                                const float* __restrict__ S,
                                                float* __restrict__ out) {
    __shared__ unsigned lQ[128 * 16];  // 128 rows x 32 bf16
    __shared__ unsigned lS[64 * 32];   // transposed: [e][64 d] bf16
    int mc = blockIdx.x, h = blockIdx.y, b = blockIdx.z;
    int tid = threadIdx.x, wid = tid >> 5, lane = tid & 31;
    const float* Qp = Q + ((size_t)b * 1024 + mc * 128) * 1024 + h * 64;
    const float* Sp = S + ((size_t)(b * 16 + h)) * 64 * 64;

    {
        int d  = tid >> 2;
        int eg = (tid & 3) * 16;
        unsigned short* sS = (unsigned short*)lS;
        for (int i = 0; i < 4; ++i) {
            float4 f = *(const float4*)(Sp + (size_t)d * 64 + eg + 4 * i);
            sS[(eg + 4 * i + 0) * 64 + d] = bf16h(f.x);
            sS[(eg + 4 * i + 1) * 64 + d] = bf16h(f.y);
            sS[(eg + 4 * i + 2) * 64 + d] = bf16h(f.z);
            sS[(eg + 4 * i + 3) * 64 + d] = bf16h(f.w);
        }
    }

    const int srow = tid >> 1, shalf = tid & 1;
    const int wm = wid * 16;
    v8f acc[4] = {};

    for (int ks = 0; ks < 2; ++ks) {            // K = 64 = 2 x 32
        __syncthreads();
        pack8(&lQ[srow * 16 + shalf * 8],
              Qp + (size_t)srow * 1024 + ks * 32 + shalf * 16);
        __syncthreads();

        const int abase = (lane >> 4) << 2;
        const unsigned* pA = &lQ[(wm + (lane & 15)) * 16];
        v8u a;
        a[0] = pA[abase + 0]; a[1] = pA[abase + 1];
        a[2] = pA[abase + 2]; a[3] = pA[abase + 3];
        a[4] = pA[abase + 8]; a[5] = pA[abase + 9];
        a[6] = pA[abase + 10]; a[7] = pA[abase + 11];
        v16bf af = __builtin_bit_cast(v16bf, a);

        for (int u = 0; u < 4; ++u) {
            int e = u * 16 + (lane & 15);
            int bbase = e * 32 + ks * 16 + ((lane >> 4) << 2);
            v8u bb;
            for (int j = 0; j < 8; ++j) bb[j] = lS[bbase + j + (j & 4)];
            acc[u] = wmma_bf16(af, __builtin_bit_cast(v16bf, bb), acc[u]);
        }
    }

    int rbase = mc * 128 + wm + ((lane >> 4) << 3);
    for (int u = 0; u < 4; ++u) {
        int e = h * 64 + u * 16 + (lane & 15);
        for (int r = 0; r < 8; ++r)
            out[((size_t)b * 1024 + rbase + r) * 1024 + e] = acc[u][r];
    }
}

// ---------------- host launch ----------------
extern "C" void kernel_launch(void* const* d_in, const int* in_sizes, int n_in,
                              void* d_out, int out_size, void* d_ws, size_t ws_size,
                              hipStream_t stream) {
    const float* qimg = (const float*)d_in[0];
    const float* kimg = (const float*)d_in[1];
    const float* vimg = (const float*)d_in[2];
    const float* wq = (const float*)d_in[3];
    const float* bq = (const float*)d_in[4];
    const float* wk = (const float*)d_in[5];
    const float* bk = (const float*)d_in[6];
    const float* wv = (const float*)d_in[7];
    const float* bv = (const float*)d_in[8];

    float* ws = (float*)d_ws;
    float* pe = ws;                                  // 1024*1024 f32
    float* Tq = pe + (size_t)1024 * 1024;            // 16M f32 each
    float* Tk = Tq + (size_t)16 * 1024 * 1024;
    float* Tv = Tk + (size_t)16 * 1024 * 1024;
    float* S  = Tv + (size_t)16 * 1024 * 1024;       // 16*16*64*64 f32
    unsigned short* Xqb = (unsigned short*)(S + (size_t)1024 * 1024);
    unsigned short* Xkb = Xqb + (size_t)12582912;    // 16*3*512*512 bf16 each
    unsigned short* Xvb = Xkb + (size_t)12582912;
    unsigned short* Wqb = Xvb + (size_t)12582912;    // 1024*768 bf16 each
    unsigned short* Wkb = Wqb + (size_t)786432;
    unsigned short* Wvb = Wkb + (size_t)786432;

    // fp32 -> bf16 conversion passes (images + weights)
    to_bf16<<<6144, 256, 0, stream>>>(qimg, (uint4*)Xqb);
    to_bf16<<<6144, 256, 0, stream>>>(kimg, (uint4*)Xkb);
    to_bf16<<<6144, 256, 0, stream>>>(vimg, (uint4*)Xvb);
    to_bf16<<<384, 256, 0, stream>>>(wq, (uint4*)Wqb);
    to_bf16<<<384, 256, 0, stream>>>(wk, (uint4*)Wkb);
    to_bf16<<<384, 256, 0, stream>>>(wv, (uint4*)Wvb);

    pe_kernel<<<2048, 256, 0, stream>>>(pe);

    dim3 ggrid(128, 8);
    embed_gemm<<<ggrid, 256, 0, stream>>>(Xqb, Wqb, bq, pe, Tq);
    embed_gemm<<<ggrid, 256, 0, stream>>>(Xkb, Wkb, bk, pe, Tk);
    embed_gemm<<<ggrid, 256, 0, stream>>>(Xvb, Wvb, bv, pe, Tv);

    qsoftmax<<<32768, 256, 0, stream>>>(Tq);
    ksoftmax<<<dim3(4, 16), 256, 0, stream>>>(Tk);

    kv_scores<<<dim3(16, 16), 128, 0, stream>>>(Tk, Tv, S);
    ctx_gemm<<<dim3(8, 16, 16), 256, 0, stream>>>(Tq, S, (float*)d_out);
}
